// AttentionBasedGNNLayer_17575006175972
// MI455X (gfx1250) — matrix operand order repaired and checked by reading, not compiled, not run on hardware
//
#include <hip/hip_runtime.h>
#include <hip/hip_bf16.h>

// ---------------------------------------------------------------------------
// AttentionBasedGNNLayer: MHA with additive mask, MI455X (gfx1250, wave32)
// All GEMMs via v_wmma_f32_16x16x32_bf16; fp32 online softmax.
// L=2048, B=4, E=512, H=8, DH=64
// GEMM wave tile 64x32 (8 f32 accum frags = 64 VGPRs) to avoid scratch spills.
// ---------------------------------------------------------------------------

#define LSEQ 2048
#define BB   4
#define EE   512
#define HH   8
#define DHD  64
#define NTOK (LSEQ * BB)   // 8192

typedef __attribute__((ext_vector_type(16))) __bf16 v16bf;
typedef __attribute__((ext_vector_type(8)))  float  v8f;

struct __align__(16) u32x4 { unsigned int x, y, z, w; };

union Frag {                 // one WMMA 16x32 bf16 operand per lane: 8 VGPRs
    v16bf v;
    u32x4 u[2];
};

static __device__ __forceinline__ unsigned short f2bf(float f) {
    __bf16 h = (__bf16)f;                     // RNE convert
    return __builtin_bit_cast(unsigned short, h);
}

static __device__ __forceinline__ u32x4 ld16(const unsigned short* p) {
    return *(const u32x4*)p;
}

static __device__ __forceinline__ v8f wmma_bf16(v16bf a, v16bf b, v8f c) {
    // (neg_a, A, neg_b, B, c_mod, C, reuse_a, reuse_b)
    return __builtin_amdgcn_wmma_f32_16x16x32_bf16(false, a, false, b,
                                                   (short)0, c, false, false);
}

// ---------------------------------------------------------------------------
// fp32 -> bf16 elementwise convert
// ---------------------------------------------------------------------------
__global__ void cvt_f32_bf16(const float* __restrict__ in,
                             unsigned short* __restrict__ out, int n) {
    int i = blockIdx.x * blockDim.x + threadIdx.x;
    if (i < n) out[i] = f2bf(in[i]);
}

// ---------------------------------------------------------------------------
// QKV projection: C[t,e] = sum_c A[t,c] * W[e,c] + bias[e]   (NT GEMM)
// A: 8192x512 bf16 row-major (t = l*B + b), W: 512x512 bf16 row-major.
// mode 0/1: store (B,H,L,DH) layout (q is pre-scaled by 1/sqrt(DH));
// mode 2  : store V transposed (B,H,DH,L).
// Block = 256 threads (8 waves, 2x4), block tile 128x128, wave tile 64x32.
// ---------------------------------------------------------------------------
__global__ void gemm_qkv(const unsigned short* __restrict__ A,
                         const unsigned short* __restrict__ W,
                         const float* __restrict__ bias,
                         unsigned short* __restrict__ out,
                         float scale, int mode) {
    const int wave = threadIdx.x >> 5;
    const int ln   = threadIdx.x & 31;
    const int half = ln >> 4;
    const int l16  = ln & 15;
    const int m0 = blockIdx.x * 128 + (wave >> 2) * 64;   // 2 waves along M
    const int n0 = blockIdx.y * 128 + (wave & 3) * 32;    // 4 waves along N

    v8f acc[4][2];
#pragma unroll
    for (int i = 0; i < 4; ++i)
#pragma unroll
        for (int j = 0; j < 2; ++j) acc[i][j] = {};

    for (int k0 = 0; k0 < EE; k0 += 32) {
        Frag aF[4], bF[2];
#pragma unroll
        for (int i = 0; i < 4; ++i) {     // A 16x32: lane=row, K split by half
            const int row = m0 + i * 16 + l16;
            const unsigned short* p = A + row * EE + k0 + half * 8;
            aF[i].u[0] = ld16(p);
            aF[i].u[1] = ld16(p + 16);
        }
#pragma unroll
        for (int j = 0; j < 2; ++j) {     // B 32x16: lane=col, K split by half
            const int col = n0 + j * 16 + l16;
            const unsigned short* p = W + col * EE + k0 + half * 8;
            bF[j].u[0] = ld16(p);
            bF[j].u[1] = ld16(p + 16);
        }
#pragma unroll
        for (int i = 0; i < 4; ++i)
#pragma unroll
            for (int j = 0; j < 2; ++j)
                acc[i][j] = wmma_bf16(aF[i].v, bF[j].v, acc[i][j]);
    }

#pragma unroll
    for (int i = 0; i < 4; ++i)
#pragma unroll
        for (int j = 0; j < 2; ++j)
#pragma unroll
            for (int r = 0; r < 8; ++r) {
                const int row = m0 + i * 16 + r + half * 8;  // token t
                const int col = n0 + j * 16 + l16;           // feature e
                const float val = (acc[i][j][r] + bias[col]) * scale;
                const int l = row >> 2, b = row & 3;         // t = l*B + b
                const int h = col >> 6, d = col & 63;        // e = h*DH + d
                size_t idx;
                if (mode == 2)
                    idx = ((size_t)(b * HH + h) * DHD + d) * LSEQ + l;
                else
                    idx = ((size_t)(b * HH + h) * LSEQ + l) * DHD + d;
                out[idx] = f2bf(val);
            }
}

// ---------------------------------------------------------------------------
// Flash-style attention. Grid (L/64, B*H), block 128 (4 waves).
// Each wave: 16 query rows; streams 32 key columns per iteration.
// S = Q*K^T (+adj), fp32 online softmax, ctx += P*V (P via LDS relayout).
// ---------------------------------------------------------------------------
__global__ void attn_flash(const unsigned short* __restrict__ qh,  // (B,H,L,DH) pre-scaled
                           const unsigned short* __restrict__ kh,  // (B,H,L,DH)
                           const unsigned short* __restrict__ vt,  // (B,H,DH,L)
                           const float* __restrict__ adj,          // (L,L)
                           unsigned short* __restrict__ ctx) {     // (L,B,E) bf16
    __shared__ __align__(16) unsigned short ldsP[4][16 * 32];  // per-wave P tile

    const int wave = threadIdx.x >> 5;
    const int ln   = threadIdx.x & 31;
    const int half = ln >> 4;
    const int l16  = ln & 15;
    const int bh = blockIdx.y;             // b*H + h
    const int b = bh >> 3, h = bh & 7;
    const int r0 = blockIdx.x * 64 + wave * 16;

    const unsigned short* qbase = qh + (size_t)bh * LSEQ * DHD;
    const unsigned short* kbase = kh + (size_t)bh * LSEQ * DHD;
    const unsigned short* vbase = vt + (size_t)bh * DHD * LSEQ;

    // Q rows for this wave: two 16x32 A-fragments (DH = 64)
    Frag aQ[2];
    {
        const int row = r0 + l16;
        const int koff = half * 8;
#pragma unroll
        for (int kk = 0; kk < 2; ++kk) {
            const unsigned short* p = qbase + row * DHD + kk * 32 + koff;
            aQ[kk].u[0] = ld16(p);
            aQ[kk].u[1] = ld16(p + 16);
        }
    }

    float m_run[8], s_run[8];
    v8f O[4];
#pragma unroll
    for (int r = 0; r < 8; ++r) { m_run[r] = -1e30f; s_run[r] = 0.0f; }
#pragma unroll
    for (int f = 0; f < 4; ++f) O[f] = {};

    unsigned short* lp = ldsP[wave];

    for (int jt = 0; jt < LSEQ / 32; ++jt) {
        const int j0 = jt * 32;

        // ---- S = Q * K^T over 32 key columns (two 16x16 accumulators) ----
        v8f accS[2];
#pragma unroll
        for (int jj = 0; jj < 2; ++jj) {
            accS[jj] = {};
            const int tok = j0 + jj * 16 + l16;
            const int coff = half * 8;
#pragma unroll
            for (int kk = 0; kk < 2; ++kk) {
                Frag bK;
                const unsigned short* p = kbase + tok * DHD + kk * 32 + coff;
                bK.u[0] = ld16(p);
                bK.u[1] = ld16(p + 16);
                accS[jj] = wmma_bf16(aQ[kk].v, bK.v, accS[jj]);
            }
        }

        // ---- add adjacency mask (fp32) ----
        float sv[2][8];
#pragma unroll
        for (int jj = 0; jj < 2; ++jj)
#pragma unroll
            for (int r = 0; r < 8; ++r) {
                const int rowg = r0 + r + half * 8;
                const int colg = j0 + jj * 16 + l16;
                sv[jj][r] = accS[jj][r] + adj[rowg * LSEQ + colg];
            }

        // ---- online softmax update (row spans 16 lanes of one half) ----
        float fac[8];
#pragma unroll
        for (int r = 0; r < 8; ++r) {
            float mx = fmaxf(sv[0][r], sv[1][r]);
            for (int msk = 1; msk <= 8; msk <<= 1)
                mx = fmaxf(mx, __shfl_xor(mx, msk, 32));
            const float newm = fmaxf(m_run[r], mx);
            fac[r] = __expf(m_run[r] - newm);
            const float p0 = __expf(sv[0][r] - newm);
            const float p1 = __expf(sv[1][r] - newm);
            sv[0][r] = p0; sv[1][r] = p1;
            float ps = p0 + p1;
            for (int msk = 1; msk <= 8; msk <<= 1)
                ps += __shfl_xor(ps, msk, 32);
            s_run[r] = s_run[r] * fac[r] + ps;
            m_run[r] = newm;
        }

        // ---- rescale running context ----
#pragma unroll
        for (int f = 0; f < 4; ++f)
#pragma unroll
            for (int r = 0; r < 8; ++r) O[f][r] *= fac[r];

        // ---- P (16x32 bf16) C-layout -> A-layout via private LDS bounce ----
#pragma unroll
        for (int jj = 0; jj < 2; ++jj)
#pragma unroll
            for (int r = 0; r < 8; ++r) {
                const int row = r + half * 8;
                const int col = jj * 16 + l16;
                lp[row * 32 + col] = f2bf(sv[jj][r]);
            }
        Frag aP;
        {
            const int koff = half * 8;
            const unsigned short* p = lp + l16 * 32 + koff;
            aP.u[0] = *(const u32x4*)(p);
            aP.u[1] = *(const u32x4*)(p + 16);
        }

        // ---- O += P * V (V transposed: contiguous along sequence) ----
#pragma unroll
        for (int f = 0; f < 4; ++f) {
            Frag bV;
            const int d = f * 16 + l16;
            const unsigned short* p = vbase + d * LSEQ + j0 + half * 8;
            bV.u[0] = ld16(p);
            bV.u[1] = ld16(p + 16);
            O[f] = wmma_bf16(aP.v, bV.v, O[f]);
        }
    }

    // ---- finalize: divide by softmax sum, store ctx (L,B,E) bf16 ----
#pragma unroll
    for (int r = 0; r < 8; ++r) s_run[r] = 1.0f / s_run[r];
#pragma unroll
    for (int f = 0; f < 4; ++f)
#pragma unroll
        for (int r = 0; r < 8; ++r) {
            const int lrow = r0 + r + half * 8;   // sequence position
            const int dcol = f * 16 + l16;        // dim within head
            const float val = O[f][r] * s_run[r];
            ctx[(size_t)(lrow * BB + b) * EE + h * DHD + dcol] = f2bf(val);
        }
}

// ---------------------------------------------------------------------------
// Output projection: out[t,e] = sum_c ctx[t,c]*Wo[e,c] + bo[e]  (fp32 out)
// Same 64x32 wave tile / 256-thread block as gemm_qkv.
// ---------------------------------------------------------------------------
__global__ void gemm_out(const unsigned short* __restrict__ A,   // ctx bf16 8192x512
                         const unsigned short* __restrict__ W,   // Wo bf16 512x512
                         const float* __restrict__ bias,
                         float* __restrict__ out) {
    const int wave = threadIdx.x >> 5;
    const int ln   = threadIdx.x & 31;
    const int half = ln >> 4;
    const int l16  = ln & 15;
    const int m0 = blockIdx.x * 128 + (wave >> 2) * 64;
    const int n0 = blockIdx.y * 128 + (wave & 3) * 32;

    v8f acc[4][2];
#pragma unroll
    for (int i = 0; i < 4; ++i)
#pragma unroll
        for (int j = 0; j < 2; ++j) acc[i][j] = {};

    for (int k0 = 0; k0 < EE; k0 += 32) {
        Frag aF[4], bF[2];
#pragma unroll
        for (int i = 0; i < 4; ++i) {
            const int row = m0 + i * 16 + l16;
            const unsigned short* p = A + row * EE + k0 + half * 8;
            aF[i].u[0] = ld16(p);
            aF[i].u[1] = ld16(p + 16);
        }
#pragma unroll
        for (int j = 0; j < 2; ++j) {
            const int col = n0 + j * 16 + l16;
            const unsigned short* p = W + col * EE + k0 + half * 8;
            bF[j].u[0] = ld16(p);
            bF[j].u[1] = ld16(p + 16);
        }
#pragma unroll
        for (int i = 0; i < 4; ++i)
#pragma unroll
            for (int j = 0; j < 2; ++j)
                acc[i][j] = wmma_bf16(aF[i].v, bF[j].v, acc[i][j]);
    }

#pragma unroll
    for (int i = 0; i < 4; ++i)
#pragma unroll
        for (int j = 0; j < 2; ++j)
#pragma unroll
            for (int r = 0; r < 8; ++r) {
                const int row = m0 + i * 16 + r + half * 8;
                const int col = n0 + j * 16 + l16;
                out[(size_t)row * EE + col] = acc[i][j][r] + bias[col];
            }
}

// ---------------------------------------------------------------------------
// Workspace layout (bytes)
// ---------------------------------------------------------------------------
#define OFF_XH   ((size_t)0)          // 8192*512*2  = 8388608
#define OFF_WQH  ((size_t)8388608)    // 512*512*2   = 524288
#define OFF_WKH  ((size_t)8912896)
#define OFF_WVH  ((size_t)9437184)
#define OFF_WOH  ((size_t)9961472)
#define OFF_QH   ((size_t)10485760)   // 8192*512*2
#define OFF_KH   ((size_t)18874368)
#define OFF_VT   ((size_t)27262976)
#define OFF_CTX  ((size_t)35651584)   // end = 44040192

extern "C" void kernel_launch(void* const* d_in, const int* in_sizes, int n_in,
                              void* d_out, int out_size, void* d_ws, size_t ws_size,
                              hipStream_t stream) {
    const float* x   = (const float*)d_in[0];
    const float* adj = (const float*)d_in[1];
    const float* Wq  = (const float*)d_in[2];
    const float* bq  = (const float*)d_in[3];
    const float* Wk  = (const float*)d_in[4];
    const float* bk  = (const float*)d_in[5];
    const float* Wv  = (const float*)d_in[6];
    const float* bv  = (const float*)d_in[7];
    const float* Wo  = (const float*)d_in[8];
    const float* bo  = (const float*)d_in[9];
    float* out = (float*)d_out;

    char* ws = (char*)d_ws;
    unsigned short* xh   = (unsigned short*)(ws + OFF_XH);
    unsigned short* Wqh  = (unsigned short*)(ws + OFF_WQH);
    unsigned short* Wkh  = (unsigned short*)(ws + OFF_WKH);
    unsigned short* Wvh  = (unsigned short*)(ws + OFF_WVH);
    unsigned short* Woh  = (unsigned short*)(ws + OFF_WOH);
    unsigned short* qh   = (unsigned short*)(ws + OFF_QH);
    unsigned short* kh   = (unsigned short*)(ws + OFF_KH);
    unsigned short* vt   = (unsigned short*)(ws + OFF_VT);
    unsigned short* ctxh = (unsigned short*)(ws + OFF_CTX);

    const int nX = NTOK * EE;      // 4194304
    const int nW = EE * EE;        // 262144
    cvt_f32_bf16<<<(nX + 255) / 256, 256, 0, stream>>>(x,  xh,  nX);
    cvt_f32_bf16<<<(nW + 255) / 256, 256, 0, stream>>>(Wq, Wqh, nW);
    cvt_f32_bf16<<<(nW + 255) / 256, 256, 0, stream>>>(Wk, Wkh, nW);
    cvt_f32_bf16<<<(nW + 255) / 256, 256, 0, stream>>>(Wv, Wvh, nW);
    cvt_f32_bf16<<<(nW + 255) / 256, 256, 0, stream>>>(Wo, Woh, nW);

    dim3 gG(NTOK / 128, EE / 128);   // (64, 4)
    dim3 bG(256);
    gemm_qkv<<<gG, bG, 0, stream>>>(xh, Wqh, bq, qh, 0.125f, 0);  // 1/sqrt(64)
    gemm_qkv<<<gG, bG, 0, stream>>>(xh, Wkh, bk, kh, 1.0f, 1);
    gemm_qkv<<<gG, bG, 0, stream>>>(xh, Wvh, bv, vt, 1.0f, 2);

    attn_flash<<<dim3(LSEQ / 64, BB * HH), 128, 0, stream>>>(qh, kh, vt, adj, ctxh);

    gemm_out<<<gG, bG, 0, stream>>>(ctxh, Woh, bo, out);
}
